// Attention_dec_32461362823500
// MI455X (gfx1250) — matrix operand order repaired and checked by reading, not compiled, not run on hardware
//
#include <hip/hip_runtime.h>

typedef __attribute__((ext_vector_type(16))) _Float16 v16h;
typedef __attribute__((ext_vector_type(8)))  _Float16 v8h;
typedef __attribute__((ext_vector_type(8)))  float    v8f;

union U16 { v16h v; v8h h[2]; _Float16 e[16]; };
union UF8 { v8f v; float e[8]; };

__device__ __forceinline__ v8f wmma_f16(v16h a, v16h b, v8f c) {
  // D = A(16x32 f16) x B(32x16 f16) + C(16x16 f32)
  return __builtin_amdgcn_wmma_f32_16x16x32_f16(false, a, false, b, (short)0, c, false, false);
}

// ---------------------------------------------------------------- prep kernels

__global__ __launch_bounds__(256) void cvt_f32_to_f16(_Float16* __restrict__ dst,
                                                      const float* __restrict__ src, int n) {
  int i = blockIdx.x * blockDim.x + threadIdx.x;
  if (i < n) dst[i] = (_Float16)src[i];
}

// im2col for conv1 (k=2, s=2): Ph[p, c*4 + i*2 + j] = x[(2*i2+i)*64 + 2*j2+j, c], p = i2*32+j2
__global__ __launch_bounds__(256) void im2col_k(_Float16* __restrict__ Ph,
                                                const float* __restrict__ x) {
  int idx = blockIdx.x * blockDim.x + threadIdx.x;
  if (idx >= 1024 * 512) return;
  int p = idx >> 9, q = idx & 511;
  int c = q >> 2, i = (q >> 1) & 1, j = q & 1;
  int i2 = p >> 5, j2 = p & 31;
  int n = (2 * i2 + i) * 64 + (2 * j2 + j);
  Ph[idx] = (_Float16)x[n * 128 + c];
}

// conv2 after up2 collapses to 1x1 conv: W2sum[o,c] = sum_{i,j} w2[o,c,i,j]
__global__ __launch_bounds__(256) void w2sum_k(_Float16* __restrict__ W2h,
                                               const float* __restrict__ w2) {
  int idx = blockIdx.x * blockDim.x + threadIdx.x;
  if (idx >= 128 * 128) return;
  const float* s = w2 + (size_t)idx * 4;
  W2h[idx] = (_Float16)(s[0] + s[1] + s[2] + s[3]);
}

// ---------------------------------------------------------------- generic WMMA GEMM
// C[M,N] = A[M,K](f16, row-major) x Bt[N,K]^T (f16, row-major = torch weight) + bias
// mode 0: store f16 to Cout
// mode 1: store f16 to Cout; cols >=128 additionally transposed into Vt[(n-128)*M + r]
// mode 2: f32 store with nearest-neighbor 2x up-broadcast into Fout[4096, N]
__global__ __launch_bounds__(128) void gemm_wmma_k(const _Float16* __restrict__ A,
                                                   const _Float16* __restrict__ Bt,
                                                   _Float16* __restrict__ Cout,
                                                   const float* __restrict__ bias,
                                                   int M, int N, int K, int mode,
                                                   _Float16* __restrict__ Vt,
                                                   float* __restrict__ Fout) {
  int wv = blockIdx.x * (blockDim.x >> 5) + (threadIdx.x >> 5);
  int tiles_n = N >> 4;
  int tiles = (M >> 4) * tiles_n;
  if (wv >= tiles) return;
  int tm = wv / tiles_n, tn = wv % tiles_n;
  int l = threadIdx.x & 31;
  int half = l >> 4, lr = l & 15;

  UF8 acc;
#pragma unroll
  for (int e = 0; e < 8; e++) acc.e[e] = 0.f;

  // A frag: lane row = lr; K run kb..kb+15, kb = half*16
  const _Float16* arow = A + (size_t)(tm * 16 + lr) * K + half * 16;
  // Bt frag: lane col = lr; K run kb..kb+15, kb = half*16  (same k order as A)
  const _Float16* brow = Bt + (size_t)(tn * 16 + lr) * K + half * 16;

  for (int k0 = 0; k0 < K; k0 += 32) {
    U16 a, b;
    a.v = *(const v16h*)(arow + k0);
    b.v = *(const v16h*)(brow + k0);
    acc.v = wmma_f16(a.v, b.v, acc.v);
  }

  int n = tn * 16 + lr;
  float bv = bias ? bias[n] : 0.f;
#pragma unroll
  for (int e = 0; e < 8; e++) {
    int r = tm * 16 + e + half * 8;          // C layout: lane col=n, rows M=e(+8)
    float val = acc.e[e] + bv;
    if (mode == 2) {
      int i2 = r >> 5, j2 = r & 31;          // r = i2*32+j2 (low-res pixel)
#pragma unroll
      for (int dy = 0; dy < 2; dy++)
#pragma unroll
        for (int dx = 0; dx < 2; dx++)
          Fout[(size_t)((2 * i2 + dy) * 64 + (2 * j2 + dx)) * N + n] = val;
    } else {
      Cout[(size_t)r * N + n] = (_Float16)val;
      if (mode == 1 && n >= 128) Vt[(size_t)(n - 128) * M + r] = (_Float16)val;
    }
  }
}

// ---------------------------------------------------------------- flash attention
// Q' [1024,128] (1024 distinct queries), KV [4096,256] (K|V), Vt [128][4096] = V^T per head.
// One wave per (head, 16-query tile): online-softmax over 4096 keys in 32-key blocks.
__global__ __launch_bounds__(128) void attn_flash_k(const _Float16* __restrict__ Qh,
                                                    const _Float16* __restrict__ KV,
                                                    const _Float16* __restrict__ Vt,
                                                    _Float16* __restrict__ Obuf) {
  __shared__ _Float16 Pt[4][16][32];           // per-wave P staging (C-layout -> A-layout)
  int l = threadIdx.x & 31;
  int wslot = threadIdx.x >> 5;
  int wid = blockIdx.x * 4 + wslot;
  if (wid >= 512) return;
  int head = wid >> 6, qt = wid & 63;
  int half = l >> 4, lr = l & 15;
  const float scale = 0.25f;                   // 1/sqrt(16)

  // Q A-frag: head-dim 16 padded to K=32 (upper half zero)
  U16 aq;
  {
    const _Float16* qrow = Qh + (size_t)(qt * 16 + lr) * 128 + head * 16;
    if (half == 0) aq.v = *(const v16h*)qrow;
    else {
#pragma unroll
    for (int e = 0; e < 16; e++) aq.e[e] = (_Float16)0.f;
    }
  }

  float m[8], lse[8];
  UF8 acc;
#pragma unroll
  for (int e = 0; e < 8; e++) { m[e] = -1e30f; lse[e] = 0.f; acc.e[e] = 0.f; }

  const _Float16* kp = KV + head * 16;
  const _Float16* vp = Vt + (size_t)(head * 16 + lr) * 4096 + half * 16;

  for (int kb = 0; kb < 4096; kb += 32) {
    // K^T B-frags (per-head rows of KV are contiguous); K-dim 16..31 is zero padding
    U16 b0, b1;
    if (half == 0) {
      b0.v = *(const v16h*)(kp + (size_t)(kb + lr) * 256);
      b1.v = *(const v16h*)(kp + (size_t)(kb + 16 + lr) * 256);
      if (kb + 32 < 4096) {
        __builtin_prefetch(kp + (size_t)(kb + 32 + lr) * 256, 0, 0);
        __builtin_prefetch(kp + (size_t)(kb + 48 + lr) * 256, 0, 0);
      }
    } else {
#pragma unroll
      for (int e = 0; e < 16; e++) { b0.e[e] = (_Float16)0.f; b1.e[e] = (_Float16)0.f; }
    }

    v8f z = {};
    UF8 s0, s1;
    s0.v = wmma_f16(aq.v, b0.v, z);            // scores for keys kb..kb+15
    s1.v = wmma_f16(aq.v, b1.v, z);            // scores for keys kb+16..kb+31

    float rmax[8];
#pragma unroll
    for (int e = 0; e < 8; e++) {
      s0.e[e] *= scale; s1.e[e] *= scale;
      rmax[e] = fmaxf(s0.e[e], s1.e[e]);
    }
#pragma unroll
    for (int off = 1; off < 16; off <<= 1)     // row max across the 16-lane half-group
#pragma unroll
      for (int e = 0; e < 8; e++) rmax[e] = fmaxf(rmax[e], __shfl_xor(rmax[e], off, 32));

    float rs[8];
#pragma unroll
    for (int e = 0; e < 8; e++) {
      float mn = fmaxf(m[e], rmax[e]);
      float alpha = __expf(m[e] - mn);
      m[e] = mn;
      s0.e[e] = __expf(s0.e[e] - mn);
      s1.e[e] = __expf(s1.e[e] - mn);
      rs[e] = s0.e[e] + s1.e[e];
      acc.e[e] *= alpha;
      lse[e] *= alpha;
    }
#pragma unroll
    for (int off = 1; off < 16; off <<= 1)     // row sum across half-group
#pragma unroll
      for (int e = 0; e < 8; e++) rs[e] += __shfl_xor(rs[e], off, 32);
#pragma unroll
    for (int e = 0; e < 8; e++) lse[e] += rs[e];

    // stage P (C-layout) into LDS as [query][key] f16, reload in A-layout
#pragma unroll
    for (int e = 0; e < 8; e++) {
      int row = e + half * 8;
      Pt[wslot][row][lr] = (_Float16)s0.e[e];
      Pt[wslot][row][16 + lr] = (_Float16)s1.e[e];
    }
    asm volatile("s_wait_dscnt 0x0" ::: "memory");   // RAW: stores -> A-frag loads

    U16 ap;
    const _Float16* prow = &Pt[wslot][lr][half * 16];
    ap.v = *(const v16h*)prow;                 // keys kbase..kbase+15

    U16 bvf;                                   // V B-frag: contiguous per-head V^T row
    bvf.v = *(const v16h*)(vp + kb);

    acc.v = wmma_f16(ap.v, bvf.v, acc.v);
    asm volatile("s_wait_dscnt 0x0" ::: "memory");   // WAR: loads done before next-iter stores
  }

#pragma unroll
  for (int e = 0; e < 8; e++) {
    int row = e + half * 8;
    Obuf[(size_t)(qt * 16 + row) * 128 + head * 16 + lr] = (_Float16)(acc.e[e] / lse[e]);
  }
}

// ---------------------------------------------------------------- launch

extern "C" void kernel_launch(void* const* d_in, const int* in_sizes, int n_in,
                              void* d_out, int out_size, void* d_ws, size_t ws_size,
                              hipStream_t stream) {
  (void)in_sizes; (void)n_in; (void)out_size; (void)ws_size;
  const float* x   = (const float*)d_in[0];
  const float* w1  = (const float*)d_in[1];
  const float* b1  = (const float*)d_in[2];
  const float* w2  = (const float*)d_in[3];
  const float* b2  = (const float*)d_in[4];
  const float* qw  = (const float*)d_in[5];
  const float* kvw = (const float*)d_in[6];
  const float* pw  = (const float*)d_in[7];
  const float* pb  = (const float*)d_in[8];
  float* out = (float*)d_out;
  char* w = (char*)d_ws;

  _Float16* Xh   = (_Float16*)(w + 0);         // [4096,128]
  _Float16* Ph   = (_Float16*)(w + 1048576);   // [1024,512] im2col patches
  _Float16* W1h  = (_Float16*)(w + 2097152);   // [128,512]
  _Float16* W2h  = (_Float16*)(w + 2228224);   // [128,128] collapsed conv2
  _Float16* QWh  = (_Float16*)(w + 2260992);   // [128,128]
  _Float16* KVWh = (_Float16*)(w + 2293760);   // [256,128]
  _Float16* PWh  = (_Float16*)(w + 2359296);   // [128,128]
  _Float16* Y1   = (_Float16*)(w + 2392064);   // [1024,128]
  _Float16* Y2   = (_Float16*)(w + 2654208);   // [1024,128]
  _Float16* Qh   = (_Float16*)(w + 2916352);   // [1024,128]
  _Float16* KV   = (_Float16*)(w + 3178496);   // [4096,256]
  _Float16* Vt   = (_Float16*)(w + 5275648);   // [128,4096] per-head V^T
  _Float16* Ob   = (_Float16*)(w + 6324224);   // [1024,128]

  cvt_f32_to_f16<<<2048, 256, 0, stream>>>(Xh, x, 524288);
  cvt_f32_to_f16<<<256, 256, 0, stream>>>(W1h, w1, 65536);
  cvt_f32_to_f16<<<64, 256, 0, stream>>>(QWh, qw, 16384);
  cvt_f32_to_f16<<<128, 256, 0, stream>>>(KVWh, kvw, 32768);
  cvt_f32_to_f16<<<64, 256, 0, stream>>>(PWh, pw, 16384);
  im2col_k<<<2048, 256, 0, stream>>>(Ph, x);
  w2sum_k<<<64, 256, 0, stream>>>(W2h, w2);

  // conv1 as GEMM: Y1 = Ph x W1^T + b1            (M=1024,N=128,K=512)
  gemm_wmma_k<<<128, 128, 0, stream>>>(Ph, W1h, Y1, b1, 1024, 128, 512, 0, nullptr, nullptr);
  // collapsed conv2: Y2 = Y1 x W2sum^T + b2       (M=1024,N=128,K=128)
  gemm_wmma_k<<<128, 128, 0, stream>>>(Y1, W2h, Y2, b2, 1024, 128, 128, 0, nullptr, nullptr);
  // query projection: Q' = Y2 x q_w^T
  gemm_wmma_k<<<128, 128, 0, stream>>>(Y2, QWh, Qh, nullptr, 1024, 128, 128, 0, nullptr, nullptr);
  // kv projection: KV = X x kv_w^T (+ V transposed into Vt)   (M=4096,N=256,K=128)
  gemm_wmma_k<<<1024, 128, 0, stream>>>(Xh, KVWh, KV, nullptr, 4096, 256, 128, 1, Vt, nullptr);
  // flash attention over 1024 distinct queries, 8 heads
  attn_flash_k<<<128, 128, 0, stream>>>(Qh, KV, Vt, Ob);
  // output projection + 2x nearest-neighbor broadcast to [4096,128] f32
  gemm_wmma_k<<<128, 128, 0, stream>>>(Ob, PWh, nullptr, pb, 1024, 128, 128, 2, nullptr, out);
}
